// SparseAutoencoder_62414464745525
// MI455X (gfx1250) — matrix operand order repaired and checked by reading, not compiled, not run on hardware
//
#include <hip/hip_runtime.h>
#include <hip/hip_bf16.h>

#define B_TOTAL 32768
#define D_DIM   512
#define H_DIM   2048
#define B_ROWS  16
#define TOPK    50

// packed-weight geometry: [ntile][ktile][lane(32)][16 bf16]
#define ENC_NT (H_DIM / 16)   // 128
#define ENC_KT (D_DIM / 32)   // 16
#define DEC_NT (D_DIM / 16)   // 32
#define DEC_KT (H_DIM / 32)   // 64
#define ENC_PACK_ELEMS (ENC_NT * ENC_KT * 32 * 16)   // 1,048,576
#define DEC_PACK_ELEMS (DEC_NT * DEC_KT * 32 * 16)   // 1,048,576
#define WS_NEEDED ((size_t)(ENC_PACK_ELEMS + DEC_PACK_ELEMS) * 2)  // 4 MB

typedef __attribute__((ext_vector_type(16))) __bf16 v16bf;
typedef __attribute__((ext_vector_type(8)))  float  v8f;
typedef __attribute__((ext_vector_type(4)))  float  v4f;

// order-preserving f32 -> u32 key (monotone increasing)
__device__ __forceinline__ unsigned f2key(float f) {
    unsigned u = __float_as_uint(f);
    return u ^ ((unsigned)((int)u >> 31) | 0x80000000u);
}
__device__ __forceinline__ float key2f(unsigned k) {
    unsigned u = (k & 0x80000000u) ? (k ^ 0x80000000u) : ~k;
    return __uint_as_float(u);
}

// ---------------------------------------------------------------------------
// One-time per launch: swizzle f32 weights into bf16 WMMA B-fragment layout.
//   B 32x16 (bf16, wave32), this code's k order (A matches): lane column n = lane&15; lanes<16 hold K 0..15,
//   lanes>=16 hold K 16..31; element e within lane = K offset.
// ---------------------------------------------------------------------------
__global__ __launch_bounds__(256)
void pack_weights_kernel(const float* __restrict__ W_enc,
                         const float* __restrict__ W_dec,
                         __bf16* __restrict__ Wp_enc,
                         __bf16* __restrict__ Wp_dec) {
    const int stride = gridDim.x * blockDim.x;
    for (int flat = blockIdx.x * blockDim.x + threadIdx.x;
         flat < ENC_PACK_ELEMS; flat += stride) {
        const int e    = flat & 15;
        const int lane = (flat >> 4) & 31;
        const int kt   = (flat >> 9) & (ENC_KT - 1);
        const int nt   = flat >> 13;
        const int h = nt * 16 + (lane & 15);
        const int k = kt * 32 + (lane >> 4) * 16 + e;
        Wp_enc[flat] = (__bf16)W_enc[(size_t)h * D_DIM + k];
    }
    for (int flat = blockIdx.x * blockDim.x + threadIdx.x;
         flat < DEC_PACK_ELEMS; flat += stride) {
        const int e    = flat & 15;
        const int lane = (flat >> 4) & 31;
        const int kt   = (flat >> 9) & (DEC_KT - 1);
        const int nt   = flat >> 15;
        const int d = nt * 16 + (lane & 15);
        const int k = kt * 32 + (lane >> 4) * 16 + e;
        Wp_dec[flat] = (__bf16)W_dec[(size_t)d * H_DIM + k];
    }
}

// ---------------------------------------------------------------------------
// Fused SAE forward: encode GEMM -> radix-select top-k -> mask -> decode
// ---------------------------------------------------------------------------
template <bool PW>
__global__ __launch_bounds__(256)
void sae_fused_kernel(const float* __restrict__ x,
                      const float* __restrict__ W_enc,
                      const float* __restrict__ b_enc,
                      const float* __restrict__ W_dec,
                      const float* __restrict__ b_dec,
                      const float* __restrict__ pre_bias,
                      const __bf16* __restrict__ Wp_enc,
                      const __bf16* __restrict__ Wp_dec,
                      float* __restrict__ out_recon,
                      float* __restrict__ out_z) {
    // 16 KB scratch: phase 1/2 = packed A-tile [kt][lane][16 bf16];
    //                phase 3   = 8 per-wave 256-bin histograms (8 KB)
    __shared__ __align__(16) unsigned char lds_scratch[ENC_KT * 32 * 16 * 2];
    __shared__ float lds_z[B_ROWS * H_DIM];     // 128 KB: z_pre, then masked z
    __shared__ float lds_th[B_ROWS];

    __bf16* lds_a = reinterpret_cast<__bf16*>(lds_scratch);

    const int tid  = threadIdx.x;
    const int lane = tid & 31;
    const int wave = tid >> 5;
    const int row0 = blockIdx.x * B_ROWS;

    // Warm L2 with weight slabs (global_prefetch_b8 on gfx1250)
    if (PW) {
        __builtin_prefetch(Wp_enc + (size_t)wave * (ENC_PACK_ELEMS / 8), 0, 2);
        __builtin_prefetch(Wp_dec + (size_t)wave * (DEC_PACK_ELEMS / 8), 0, 2);
    } else {
        __builtin_prefetch(W_enc + (size_t)wave * (H_DIM * D_DIM / 8), 0, 2);
        __builtin_prefetch(W_dec + (size_t)wave * (D_DIM * H_DIM / 8), 0, 2);
    }

    // ---- Phase 1: xc = x - b_dec - pre_bias -> LDS (bf16, A-fragment order)
    // A 16x32 (bf16, wave32), this code's k order (B matches): lanes 0-15 row m hold K 0..15;
    //                         lanes 16-31 hold K 16..31.
    #pragma unroll
    for (int i = 0; i < (ENC_KT * 32 * 16) / (256 * 2); ++i) {
        const int p  = i * 256 + tid;            // pair index (k, k+1)
        const int e2 = (p & 7) * 2;
        const int ln = (p >> 3) & 31;
        const int kt = p >> 8;
        const int m  = ln & 15;
        const int ko = (ln >> 4) * 16;
        const int k  = kt * 32 + ko + e2;
        const float2 xv = *reinterpret_cast<const float2*>(&x[(size_t)(row0 + m) * D_DIM + k]);
        const float2 bd = *reinterpret_cast<const float2*>(&b_dec[k]);
        const float2 pb = *reinterpret_cast<const float2*>(&pre_bias[k]);
        lds_a[p * 2]     = (__bf16)(xv.x - bd.x - pb.x);
        lds_a[p * 2 + 1] = (__bf16)(xv.y - bd.y - pb.y);
    }
    __syncthreads();

    const int koff  = (lane >> 4) * 16;   // A-frag K sub-offset (decode/fallback)
    const int bhalf = (lane >> 4) * 16;   // B-frag K half offset (fallback)
    const int ncol  = lane & 15;          // B/C/D column within tile
    const int mrow  = lane & 15;          // A row within tile

    // ---- Phase 2: z_pre = xc @ W_enc^T + b_enc  (bf16 WMMA, f32 accum) -----
    for (int chunk = 0; chunk < 4; ++chunk) {
        const int nt0 = wave * 16 + chunk * 4;   // 8 waves x 16 tiles = 128
        v8f acc[4] = {};
        for (int kt = 0; kt < ENC_KT; ++kt) {
            const v16bf a = *reinterpret_cast<const v16bf*>(
                &lds_a[(kt * 32 + lane) * 16]);
            #pragma unroll
            for (int j = 0; j < 4; ++j) {
                const int nt = nt0 + j;
                v16bf b;
                if constexpr (PW) {
                    b = *reinterpret_cast<const v16bf*>(
                        &Wp_enc[(size_t)((nt * ENC_KT + kt) * 32 + lane) * 16]);
                } else {
                    const int h = nt * 16 + ncol;
                    const float* brow = &W_enc[(size_t)h * D_DIM + kt * 32 + bhalf];
                    #pragma unroll
                    for (int q = 0; q < 4; ++q) {
                        float4 pq = *reinterpret_cast<const float4*>(brow + 4 * q);
                        b[4 * q]     = (__bf16)pq.x;
                        b[4 * q + 1] = (__bf16)pq.y;
                        b[4 * q + 2] = (__bf16)pq.z;
                        b[4 * q + 3] = (__bf16)pq.w;
                    }
                }
                acc[j] = __builtin_amdgcn_wmma_f32_16x16x32_bf16(
                    false, a, false, b, (short)0, acc[j], false, false);
            }
        }
        #pragma unroll
        for (int j = 0; j < 4; ++j) {
            const int h = (nt0 + j) * 16 + ncol;
            const float be = b_enc[h];
            #pragma unroll
            for (int r = 0; r < 8; ++r) {
                const int m = r + (lane >> 4) * 8;   // C/D layout
                lds_z[m * H_DIM + h] = acc[j][r] + be;
            }
        }
    }
    __syncthreads();

    // ---- Phase 3: exact k-th largest per row via 4x8-bit radix select ------
    {
        unsigned* hist = reinterpret_cast<unsigned*>(lds_scratch) + wave * 256;
        for (int rr = 0; rr < 2; ++rr) {
            const int row = wave * 2 + rr;            // 8 waves x 2 rows = 16
            const float* zr = &lds_z[row * H_DIM];
            unsigned prefixKey = 0u;
            int kneed = TOPK;
            #pragma unroll
            for (int pass = 0; pass < 4; ++pass) {
                const int sh = 24 - 8 * pass;
                const unsigned maskHi =
                    (pass == 0) ? 0u : (0xFFFFFFFFu << (sh + 8));
                // zero this wave's 256 bins (8 per lane, contiguous)
                uint4 z4 = {0u, 0u, 0u, 0u};
                *reinterpret_cast<uint4*>(&hist[lane * 8])     = z4;
                *reinterpret_cast<uint4*>(&hist[lane * 8 + 4]) = z4;
                __threadfence_block();
                // histogram prefix-matching elements
                for (int i = 0; i < H_DIM / 32; ++i) {
                    const unsigned key = f2key(zr[lane + i * 32]);
                    if (((key ^ prefixKey) & maskHi) == 0u)
                        atomicAdd(&hist[(key >> sh) & 255u], 1u);
                }
                __threadfence_block();
                // per-lane suffix sums over its 8 bins
                unsigned h[8], sj[8];
                uint4 h0 = *reinterpret_cast<const uint4*>(&hist[lane * 8]);
                uint4 h1 = *reinterpret_cast<const uint4*>(&hist[lane * 8 + 4]);
                h[0] = h0.x; h[1] = h0.y; h[2] = h0.z; h[3] = h0.w;
                h[4] = h1.x; h[5] = h1.y; h[6] = h1.z; h[7] = h1.w;
                sj[7] = h[7];
                #pragma unroll
                for (int j = 6; j >= 0; --j) sj[j] = sj[j + 1] + h[j];
                // inclusive suffix sum across lanes (lane L -> bins >= L*8)
                unsigned v = sj[0];
                #pragma unroll
                for (int off = 16; off > 0; off >>= 1) {
                    unsigned t = __shfl_down(v, off, 32);
                    if (lane + off < 32) v += t;
                }
                const unsigned above = v - sj[0];   // count in bins >= (L+1)*8
                // largest bin with (count from top) >= kneed
                int localBin = -1;
                #pragma unroll
                for (int j = 7; j >= 0; --j)
                    if (localBin < 0 && (int)(above + sj[j]) >= kneed)
                        localBin = lane * 8 + j;
                int best = localBin;
                #pragma unroll
                for (int off = 16; off > 0; off >>= 1)
                    best = max(best, __shfl_xor(best, off, 32));
                // count strictly above selected bin, from owner lane
                const int jsel = best & 7;
                unsigned myAbove = above + ((jsel < 7) ? sj[jsel + 1] : 0u);
                const unsigned cntAbove = __shfl(myAbove, best >> 3, 32);
                kneed -= (int)cntAbove;
                prefixKey |= ((unsigned)(best & 255)) << sh;
            }
            if (lane == 0) lds_th[row] = key2f(prefixKey);
        }
    }
    __syncthreads();

    // ---- Phase 4: mask z in LDS (in place) + stream z to HBM (NT, b128) ----
    for (int i = 0; i < (B_ROWS * H_DIM) / (256 * 4); ++i) {
        const int flat = (i * 256 + tid) * 4;
        const int r = flat >> 11, c = flat & (H_DIM - 1);
        v4f zp = *reinterpret_cast<const v4f*>(&lds_z[flat]);
        const float th = lds_th[r];
        v4f zt;
        zt[0] = (zp[0] >= th) ? zp[0] : 0.f;
        zt[1] = (zp[1] >= th) ? zp[1] : 0.f;
        zt[2] = (zp[2] >= th) ? zp[2] : 0.f;
        zt[3] = (zp[3] >= th) ? zp[3] : 0.f;
        *reinterpret_cast<v4f*>(&lds_z[flat]) = zt;
        __builtin_nontemporal_store(
            zt, reinterpret_cast<v4f*>(&out_z[(size_t)(row0 + r) * H_DIM + c]));
    }
    __syncthreads();

    // ---- Phase 5: recon = z @ W_dec^T + b_dec  (bf16 WMMA, f32 accum) ------
    {
        v8f acc[4] = {};                              // 8 waves x 4 tiles = 32
        for (int kt = 0; kt < DEC_KT; ++kt) {
            v16bf a;
            const float* arow = &lds_z[mrow * H_DIM + kt * 32];
            #pragma unroll
            for (int v = 0; v < 4; ++v) {
                float2 p = *reinterpret_cast<const float2*>(arow + koff + 2 * v);
                a[2 * v]     = (__bf16)p.x;
                a[2 * v + 1] = (__bf16)p.y;
            }
            #pragma unroll
            for (int v = 0; v < 4; ++v) {
                float2 p = *reinterpret_cast<const float2*>(arow + 8 + koff + 2 * v);
                a[8 + 2 * v]     = (__bf16)p.x;
                a[8 + 2 * v + 1] = (__bf16)p.y;
            }
            #pragma unroll
            for (int j = 0; j < 4; ++j) {
                const int nt = wave * 4 + j;
                v16bf b;
                if constexpr (PW) {
                    b = *reinterpret_cast<const v16bf*>(
                        &Wp_dec[(size_t)((nt * DEC_KT + kt) * 32 + lane) * 16]);
                } else {
                    const int d = nt * 16 + ncol;
                    const float* brow = &W_dec[(size_t)d * H_DIM + kt * 32 + bhalf];
                    #pragma unroll
                    for (int q = 0; q < 4; ++q) {
                        float4 pq = *reinterpret_cast<const float4*>(brow + 4 * q);
                        b[4 * q]     = (__bf16)pq.x;
                        b[4 * q + 1] = (__bf16)pq.y;
                        b[4 * q + 2] = (__bf16)pq.z;
                        b[4 * q + 3] = (__bf16)pq.w;
                    }
                }
                acc[j] = __builtin_amdgcn_wmma_f32_16x16x32_bf16(
                    false, a, false, b, (short)0, acc[j], false, false);
            }
        }
        #pragma unroll
        for (int j = 0; j < 4; ++j) {
            const int d = (wave * 4 + j) * 16 + ncol;
            const float bd = b_dec[d];
            #pragma unroll
            for (int r = 0; r < 8; ++r) {
                const int m = r + (lane >> 4) * 8;
                __builtin_nontemporal_store(
                    acc[j][r] + bd, &out_recon[(size_t)(row0 + m) * D_DIM + d]);
            }
        }
    }
}

extern "C" void kernel_launch(void* const* d_in, const int* in_sizes, int n_in,
                              void* d_out, int out_size, void* d_ws, size_t ws_size,
                              hipStream_t stream) {
    (void)in_sizes; (void)n_in; (void)out_size;
    const float* x        = (const float*)d_in[0];
    const float* W_enc    = (const float*)d_in[1];
    const float* b_enc    = (const float*)d_in[2];
    const float* W_dec    = (const float*)d_in[3];
    const float* b_dec    = (const float*)d_in[4];
    const float* pre_bias = (const float*)d_in[5];

    float* out_recon = (float*)d_out;                                  // B*D
    float* out_z     = (float*)d_out + (size_t)B_TOTAL * D_DIM;        // B*H

    dim3 grid(B_TOTAL / B_ROWS);   // 2048 workgroups, 16 rows each
    dim3 block(256);               // 8 wave32 waves

    if (ws_size >= WS_NEEDED) {
        __bf16* Wp_enc = (__bf16*)d_ws;
        __bf16* Wp_dec = Wp_enc + ENC_PACK_ELEMS;
        pack_weights_kernel<<<dim3(2048), block, 0, stream>>>(
            W_enc, W_dec, Wp_enc, Wp_dec);
        sae_fused_kernel<true><<<grid, block, 0, stream>>>(
            x, W_enc, b_enc, W_dec, b_dec, pre_bias,
            Wp_enc, Wp_dec, out_recon, out_z);
    } else {
        sae_fused_kernel<false><<<grid, block, 0, stream>>>(
            x, W_enc, b_enc, W_dec, b_dec, pre_bias,
            nullptr, nullptr, out_recon, out_z);
    }
}
